// S4ConvBlock_38027640438874
// MI455X (gfx1250) — compile-verified, matrix operand order checked by reading, not run on hardware
//
#include <hip/hip_runtime.h>
#include <hip/hip_bf16.h>
#include <math.h>

#define B_ 8
#define L_ 2048
#define H_ 1024
#define N2_ 32
#define BL_ (B_ * L_)

// GEMM tiling
#define TILE_M 64      // a-half output rows per block (g-half mirrored at +H)
#define TILE_N 128     // tokens per block
#define KSTEP  32      // bf16 WMMA K
#define LDS_ROW_B 80   // 64B of K-data padded to 80B (bank-conflict-free)

typedef float v8f  __attribute__((ext_vector_type(8)));
typedef int   v4i  __attribute__((ext_vector_type(4)));
typedef __bf16 v16bf __attribute__((ext_vector_type(16)));
typedef __bf16 v8bf  __attribute__((ext_vector_type(8)));

union BFrag {
    v16bf f;
    v8bf  h[2];
};

#if __has_builtin(__builtin_amdgcn_global_load_async_to_lds_b128)
#define USE_ASYNC_LDS 1
typedef __attribute__((address_space(1))) v4i GV4;   // global 16B chunk
typedef __attribute__((address_space(3))) v4i LV4;   // LDS 16B chunk
#define ASYNC_CP(g, l) \
    __builtin_amdgcn_global_load_async_to_lds_b128((GV4*)(g), (LV4*)(l), 0, 0)
#else
#define USE_ASYNC_LDS 0
#define ASYNC_CP(g, l) ((void)0)
#endif

#if USE_ASYNC_LDS && __has_builtin(__builtin_amdgcn_s_wait_asynccnt)
#define WAIT_ASYNC_LE(n) __builtin_amdgcn_s_wait_asynccnt(n)   // n must be literal
#elif USE_ASYNC_LDS
#define WAIT_ASYNC_LE(n) asm volatile("s_wait_asynccnt " #n ::: "memory")
#else
#define WAIT_ASYNC_LE(n) ((void)0)
#endif

__device__ __forceinline__ unsigned short f32_to_bf16_rne(float v) {
    unsigned int u = __float_as_uint(v);
    unsigned int r = 0x7FFFu + ((u >> 16) & 1u);
    return (unsigned short)((u + r) >> 16);
}

// ---------------------------------------------------------------------------
// Kernel 1: per-(h,n) SSM parameters: r = exp(dt*A), Ck = C*(exp(dt*A)-1)/A
// ---------------------------------------------------------------------------
__global__ void s4_params_kernel(const float* __restrict__ log_dt,
                                 const float* __restrict__ C_re,
                                 const float* __restrict__ C_im,
                                 const float* __restrict__ log_A_re,
                                 const float* __restrict__ A_im,
                                 float* __restrict__ rr, float* __restrict__ ri,
                                 float* __restrict__ cr, float* __restrict__ ci) {
    int i = blockIdx.x * blockDim.x + threadIdx.x;
    if (i >= H_ * N2_) return;
    int h = i >> 5;
    float dt  = expf(log_dt[h]);
    float Are = -expf(log_A_re[i]);
    float Aim = A_im[i];
    float dre = Are * dt, dim = Aim * dt;
    float e   = expf(dre);
    float r_re = e * cosf(dim);
    float r_im = e * sinf(dim);
    float nre = r_re - 1.0f, nim = r_im;
    float inv = 1.0f / (Are * Are + Aim * Aim);
    float ire = Are * inv, iim = -Aim * inv;
    float tre = nre * ire - nim * iim;
    float tim = nre * iim + nim * ire;
    float Cre = C_re[i], Cim = C_im[i];
    rr[i] = r_re;
    ri[i] = r_im;
    cr[i] = Cre * tre - Cim * tim;
    ci[i] = Cre * tim + Cim * tre;
}

// ---------------------------------------------------------------------------
// Kernel 2: W [2H,H] fp32 -> bf16
// ---------------------------------------------------------------------------
__global__ void w_to_bf16_kernel(const float* __restrict__ W,
                                 unsigned short* __restrict__ Wb, int n) {
    int i = blockIdx.x * blockDim.x + threadIdx.x;
    if (i < n) Wb[i] = f32_to_bf16_rne(W[i]);
}

// ---------------------------------------------------------------------------
// Kernel 3: diagonal-SSM scan (== FFT conv) + D skip + erf-GELU -> bf16
// ---------------------------------------------------------------------------
__global__ __launch_bounds__(256) void ssm_gelu_kernel(
        const float* __restrict__ x,
        const float* __restrict__ rr, const float* __restrict__ ri,
        const float* __restrict__ cr, const float* __restrict__ ci,
        const float* __restrict__ D,
        unsigned short* __restrict__ yb) {
    int t = blockIdx.x * blockDim.x + threadIdx.x;   // t = b*H + h
    if (t >= B_ * H_) return;
    int h = t & (H_ - 1);
    int b = t >> 10;

    float lrr[N2_], lri[N2_], lcr[N2_], lci[N2_], sre[N2_], sim[N2_];
#pragma unroll
    for (int n = 0; n < N2_; ++n) {
        int idx = h * N2_ + n;
        lrr[n] = rr[idx]; lri[n] = ri[idx];
        lcr[n] = cr[idx]; lci[n] = ci[idx];
        sre[n] = 0.0f;    sim[n] = 0.0f;
    }
    float d = D[h];
    const float*    xp = x  + (size_t)b * L_ * H_ + h;
    unsigned short* yp = yb + (size_t)b * L_ * H_ + h;

    for (int l = 0; l < L_; ++l) {
        float xv = xp[(size_t)l * H_];
        float acc = 0.0f;
#pragma unroll
        for (int n = 0; n < N2_; ++n) {
            float nr = fmaf(lrr[n], sre[n], fmaf(-lri[n], sim[n], xv));
            float ni = fmaf(lrr[n], sim[n],  lri[n] * sre[n]);
            sre[n] = nr; sim[n] = ni;
            acc = fmaf(lcr[n], nr, fmaf(-lci[n], ni, acc));
        }
        float v = 2.0f * acc + d * xv;
        float g = 0.5f * v * (1.0f + erff(v * 0.70710678118654752f));
        yp[(size_t)l * H_] = f32_to_bf16_rne(g);
    }
}

// ---------------------------------------------------------------------------
// Kernel 4: z = W @ y + b, GLU, + residual.
// Block: 256 threads (8 waves), tile 64(M) x 128(N). Wave tile: 16(M) x 64(N)
// with paired a/g accumulators (8 WMMAs per K-step). y tile staged in LDS via
// async global->LDS copies (ASYNCcnt), double-buffered, 80B-padded rows.
// ---------------------------------------------------------------------------
__global__ __launch_bounds__(256) void gemm_glu_kernel(
        const unsigned short* __restrict__ Wb_u,
        const unsigned short* __restrict__ Yb_u,
        const float* __restrict__ bias,
        const float* __restrict__ x,
        float* __restrict__ out) {
    const __bf16* Wb = (const __bf16*)Wb_u;
    const char*   Yc = (const char*)Yb_u;

    __shared__ __align__(16) char ldsY[2][TILE_N * LDS_ROW_B];   // 2 x 10240 B

    int tid  = threadIdx.x;
    int lane = tid & 31;
    int wid  = tid >> 5;
    int wm   = wid & 3;                 // 4 M-subtiles
    int wn   = wid >> 2;                // 2 N-subtiles of 64
    int h0   = blockIdx.y * TILE_M + wm * 16;
    int nblk = blockIdx.x * TILE_N;

    int lm  = lane & 15;
    int lh8 = (lane >> 4) * 8;          // K sub-offset per half-wave

    const __bf16* Arow = Wb + (size_t)(h0 + lm) * H_;    // a-half of W
    const __bf16* Grow = Arow + (size_t)H_ * H_;         // g-half of W

    // cooperative stage of one 128x32 y tile: 512 16B chunks, 2 per thread
    int c0row = tid >> 2;               // chunk tid  : rows 0..63
    int c0off = (tid & 3) << 4;
    int c1row = c0row + 64;             // chunk tid+256: rows 64..127

#define STAGE_TILE(buf, kk)                                                   \
    do {                                                                      \
        const char* g0 = Yc + (((size_t)(nblk + c0row) * H_ + (kk)) << 1) + c0off; \
        const char* g1 = Yc + (((size_t)(nblk + c1row) * H_ + (kk)) << 1) + c0off; \
        char* l0 = &ldsY[buf][c0row * LDS_ROW_B + c0off];                     \
        char* l1 = &ldsY[buf][c1row * LDS_ROW_B + c0off];                     \
        if (USE_ASYNC_LDS) {                                                  \
            ASYNC_CP(g0, l0);                                                 \
            ASYNC_CP(g1, l1);                                                 \
        } else {                                                              \
            *(v8bf*)l0 = *(const v8bf*)g0;                                    \
            *(v8bf*)l1 = *(const v8bf*)g1;                                    \
        }                                                                     \
    } while (0)

    v8f acc_a[4] = {};
    v8f acc_g[4] = {};

    STAGE_TILE(0, 0);                   // prologue: buffer 0 in flight

    for (int kt = 0; kt < H_ / KSTEP; ++kt) {
        int k   = kt * KSTEP;
        int cur = kt & 1;

        if (kt > 0) __syncthreads();    // everyone done reading buf cur^1
        if (kt + 1 < H_ / KSTEP) {
            STAGE_TILE(cur ^ 1, k + KSTEP);
            WAIT_ASYNC_LE(2);           // cur tile landed; next still in flight
        } else {
            WAIT_ASYNC_LE(0);
        }
        __syncthreads();                // whole tile visible to all waves

        // A fragments (global, L2-resident W) + prefetch ahead
        if (k + 128 < H_) {
            __builtin_prefetch(Arow + k + 128, 0, 3);
            __builtin_prefetch(Grow + k + 128, 0, 3);
        }
        BFrag fa, fg;
        fa.h[0] = *(const v8bf*)(Arow + k + lh8);
        fa.h[1] = *(const v8bf*)(Arow + k + 16 + lh8);
        fg.h[0] = *(const v8bf*)(Grow + k + lh8);
        fg.h[1] = *(const v8bf*)(Grow + k + 16 + lh8);

        const char* bbase = ldsY[cur];
#pragma unroll
        for (int s = 0; s < 4; ++s) {
            const char* rp = bbase
                + (size_t)(wn * 64 + s * 16 + lm) * LDS_ROW_B + (lh8 << 1);
            BFrag fb;
            fb.h[0] = *(const v8bf*)(rp);
            fb.h[1] = *(const v8bf*)(rp + 32);
            acc_a[s] = __builtin_amdgcn_wmma_f32_16x16x32_bf16(
                false, fa.f, false, fb.f, (short)0, acc_a[s], false, false);
            acc_g[s] = __builtin_amdgcn_wmma_f32_16x16x32_bf16(
                false, fg.f, false, fb.f, (short)0, acc_g[s], false, false);
        }
    }

    // epilogue: GLU + residual.  D layout: VGPR r -> row h0+r+8*(lane>>4),
    // lane%16 -> column n within each 16-wide N-subtile.
    int hbase = h0 + ((lane >> 4) << 3);
#pragma unroll
    for (int s = 0; s < 4; ++s) {
        int n = nblk + wn * 64 + s * 16 + lm;
        const float* xp = x   + (size_t)n * H_;
        float*       op = out + (size_t)n * H_;
#pragma unroll
        for (int r = 0; r < 8; ++r) {
            int hh = hbase + r;
            float a = acc_a[s][r] + bias[hh];
            float g = acc_g[s][r] + bias[hh + H_];
            float sg = 1.0f / (1.0f + __expf(-g));
            op[hh] = fmaf(a, sg, xp[hh]);
        }
    }
#undef STAGE_TILE
}

// ---------------------------------------------------------------------------
extern "C" void kernel_launch(void* const* d_in, const int* in_sizes, int n_in,
                              void* d_out, int out_size, void* d_ws, size_t ws_size,
                              hipStream_t stream) {
    (void)in_sizes; (void)n_in; (void)out_size; (void)ws_size;
    const float* x        = (const float*)d_in[0];
    const float* log_dt   = (const float*)d_in[1];
    const float* C_re     = (const float*)d_in[2];
    const float* C_im     = (const float*)d_in[3];
    const float* log_A_re = (const float*)d_in[4];
    const float* A_im     = (const float*)d_in[5];
    const float* D        = (const float*)d_in[6];
    const float* W        = (const float*)d_in[7];
    const float* bias     = (const float*)d_in[8];
    float* out = (float*)d_out;

    // workspace: 4 x [H*N2] f32 (512 KB) | Wb bf16 2H*H (4 MB) | y bf16 (32 MB)
    float* rr = (float*)d_ws;
    float* ri = rr + H_ * N2_;
    float* cr = ri + H_ * N2_;
    float* ci = cr + H_ * N2_;
    unsigned short* Wb = (unsigned short*)(ci + H_ * N2_);
    unsigned short* Yb = Wb + (size_t)2 * H_ * H_;

    s4_params_kernel<<<(H_ * N2_ + 255) / 256, 256, 0, stream>>>(
        log_dt, C_re, C_im, log_A_re, A_im, rr, ri, cr, ci);

    w_to_bf16_kernel<<<(2 * H_ * H_ + 255) / 256, 256, 0, stream>>>(
        W, Wb, 2 * H_ * H_);

    ssm_gelu_kernel<<<(B_ * H_ + 255) / 256, 256, 0, stream>>>(
        x, rr, ri, cr, ci, D, Yb);

    dim3 grid(BL_ / TILE_N, H_ / TILE_M);
    gemm_glu_kernel<<<grid, 256, 0, stream>>>(Wb, Yb, bias, x, out);
}
